// VectorQuantizer_15728170238286
// MI455X (gfx1250) — matrix operand order repaired and checked by reading, not compiled, not run on hardware
//
#include <hip/hip_runtime.h>

// ---------------------------------------------------------------------------
// VQ-VAE vector quantizer forward for MI455X (gfx1250, wave32, WMMA).
//   z   : [65536, 256] f32     emb : [1024, 256] f32
//   out : z_q_st [65536*256] f32  ++  loss [1] f32  ++  indices [65536] f32
// Distance argmin via  score = ||e||^2 - 2 * (z . e) with
// v_wmma_f32_16x16x32_f16; codebook tiles staged through LDS with
// double-buffered global_load_async_to_lds_b128 (ASYNCcnt pipeline).
// ---------------------------------------------------------------------------

typedef __attribute__((ext_vector_type(16))) _Float16 v16h;
typedef __attribute__((ext_vector_type(8)))  _Float16 v8h;
typedef __attribute__((ext_vector_type(8)))  float    v8f;
typedef __attribute__((ext_vector_type(4)))  float    v4f;

#define VQ_K      1024          // codebook entries
#define VQ_D      256           // embedding dim
#define VQ_ROWS   65536         // 16 * 4096
#define VQ_NCHUNK (VQ_K / 16)   // 64 chunks of 16 codes
#define VQ_KSTEP  (VQ_D / 32)   // 8 WMMA K-steps of 32 dims
#define LDS_STRIDE 264          // 256 halves + 8 pad (528B = 132 DW == 4 mod 64 banks)

// ---------------------------------------------------------------------------
// Prep: ||e_k||^2, f32 codebook -> f16 codebook, zero the loss accumulator.
// ---------------------------------------------------------------------------
__global__ __launch_bounds__(256) void vq_prep_kernel(
    const float* __restrict__ emb, _Float16* __restrict__ emb16,
    float* __restrict__ esq, float* __restrict__ loss_out) {
  const int lane = threadIdx.x & 31;
  const int wave = threadIdx.x >> 5;
  const int code = blockIdx.x * 8 + wave;

  const float* row = emb + (size_t)code * VQ_D;
  _Float16* row16  = emb16 + (size_t)code * VQ_D;

  float s = 0.0f;
#pragma unroll
  for (int t = 0; t < VQ_D / 32; ++t) {
    int d = lane + t * 32;
    float v = row[d];
    row16[d] = (_Float16)v;
    s += v * v;
  }
#pragma unroll
  for (int m = 16; m >= 1; m >>= 1) s += __shfl_xor(s, m, 32);
  if (lane == 0) esq[code] = s;

  if (blockIdx.x == 0 && threadIdx.x == 0) *loss_out = 0.0f;
}

// ---------------------------------------------------------------------------
// Main: 8 waves per block, each wave owns a 16-row block of z.
// grid = 65536 / 128 = 512 workgroups.
// ---------------------------------------------------------------------------
__global__ __launch_bounds__(256) void vq_main_kernel(
    const float* __restrict__ z, const float* __restrict__ emb,
    const _Float16* __restrict__ emb16, const float* __restrict__ esq,
    float* __restrict__ out_zq, float* __restrict__ loss_out,
    float* __restrict__ out_idx) {
  const int tid  = threadIdx.x;
  const int lane = tid & 31;
  const int wave = tid >> 5;
  const int n    = lane & 15;   // column / code-in-chunk / M-index for A loads
  const int h    = lane >> 4;   // lane half
  const int rowbase = blockIdx.x * 128 + wave * 16;

  // Double-buffered B staging: 2 x 16 codes x 264 halves (padded rows).
  __shared__ _Float16 Bbuf[2][16 * LDS_STRIDE];

  // Per-thread async-copy assignment: 8 KB chunk = 512 pieces of 16B;
  // 256 threads x 2 pieces. Piece p -> code row p>>5, 16B column p&31.
  const int p0 = tid, p1 = tid + 256;
  const int r0 = p0 >> 5, c0 = p0 & 31;
  const int r1 = p1 >> 5, c1 = p1 & 31;

#define VQ_ISSUE_CHUNK(buf, nc_)                                               \
  do {                                                                         \
    const _Float16* gsrc_ = emb16 + (size_t)(nc_) * (16 * VQ_D);               \
    unsigned l0_ = (unsigned)(unsigned long long)(const void*)                 \
                       &Bbuf[(buf)][r0 * LDS_STRIDE + c0 * 8];                 \
    unsigned l1_ = (unsigned)(unsigned long long)(const void*)                 \
                       &Bbuf[(buf)][r1 * LDS_STRIDE + c1 * 8];                 \
    const _Float16* g0_ = gsrc_ + p0 * 8;                                      \
    const _Float16* g1_ = gsrc_ + p1 * 8;                                      \
    asm volatile("global_load_async_to_lds_b128 %0, %1, off"                   \
                 :: "v"(l0_), "v"(g0_) : "memory");                            \
    asm volatile("global_load_async_to_lds_b128 %0, %1, off"                   \
                 :: "v"(l1_), "v"(g1_) : "memory");                            \
  } while (0)

  // ---- Pack the wave's A tile: 16 rows x 256 dims, f32 -> f16 fragments ----
  // 16-bit A 16x32 layout: lane (h*16 + m) holds K = h*8 + {0..7} in elements
  // 0..7 and K = 16 + h*8 + {0..7} in elements 8..15.
  v16h afrag[VQ_KSTEP];
  {
    const float* zrow = z + (size_t)(rowbase + n) * VQ_D;
#pragma unroll
    for (int kc = 0; kc < VQ_KSTEP; ++kc) {
      const float* p = zrow + kc * 32 + h * 8;
      v4f x0 = *(const v4f*)(p);
      v4f x1 = *(const v4f*)(p + 4);
      v4f y0 = *(const v4f*)(p + 16);
      v4f y1 = *(const v4f*)(p + 20);
      v16h a;
#pragma unroll
      for (int j = 0; j < 4; ++j) {
        a[j]      = (_Float16)x0[j];
        a[4 + j]  = (_Float16)x1[j];
        a[8 + j]  = (_Float16)y0[j];
        a[12 + j] = (_Float16)y1[j];
      }
      afrag[kc] = a;
    }
  }

  // Prologue of the async pipeline: land chunk 0 in buf 0.
  VQ_ISSUE_CHUNK(0, 0);
  asm volatile("s_wait_asynccnt 0x0" ::: "memory");
  __syncthreads();

  // ---- Sweep the codebook: running (min, argmin) per C-fragment register ---
  float bestv[8];
  int   besti[8];
#pragma unroll
  for (int i = 0; i < 8; ++i) { bestv[i] = 3.4e38f; besti[i] = 0; }

  int cur = 0;
  for (int nc = 0; nc < VQ_NCHUNK; ++nc) {
    // Kick the next chunk into the other buffer while we compute this one.
    if (nc + 1 < VQ_NCHUNK) VQ_ISSUE_CHUNK(cur ^ 1, nc + 1);

    const int code = nc * 16 + n;
    const float esq_n = esq[code];
    const _Float16* brow = &Bbuf[cur][n * LDS_STRIDE];

    v8f c = {0.f, 0.f, 0.f, 0.f, 0.f, 0.f, 0.f, 0.f};
#pragma unroll
    for (int kc = 0; kc < VQ_KSTEP; ++kc) {
      // 16-bit B 32x16 layout: lane (h*16 + n) holds K = h*8 + {0..7} in elements
      // 0..7 and K = 16 + h*8 + {0..7} in elements 8..15.
      const _Float16* bp = brow + kc * 32 + h * 8;
      v8h lo = *(const v8h*)(bp);
      v8h hi = *(const v8h*)(bp + 16);
      v16h b = __builtin_shufflevector(lo, hi, 0, 1, 2, 3, 4, 5, 6, 7,
                                       8, 9, 10, 11, 12, 13, 14, 15);
      c = __builtin_amdgcn_wmma_f32_16x16x32_f16(
          /*neg_a=*/false, afrag[kc], /*neg_b=*/false, b,
          /*c_mod=*/(short)0, c, /*reuse_a=*/false, /*reuse_b=*/false);
    }
#pragma unroll
    for (int i = 0; i < 8; ++i) {
      float s = fmaf(-2.0f, c[i], esq_n);  // ||e||^2 - 2 z.e
      bool better = s < bestv[i];
      bestv[i] = better ? s : bestv[i];
      besti[i] = better ? code : besti[i];
    }

    // My async copies for chunk nc+1 are done; barrier publishes the buffer
    // (and guarantees everyone is finished reading buf[cur]).
    asm volatile("s_wait_asynccnt 0x0" ::: "memory");
    __syncthreads();
    cur ^= 1;
  }

  // ---- Cross-lane argmin within each 16-lane group (row = i + 8*h) --------
#pragma unroll
  for (int i = 0; i < 8; ++i) {
#pragma unroll
    for (int m = 1; m < 16; m <<= 1) {
      float ov = __shfl_xor(bestv[i], m, 32);
      int   oi = __shfl_xor(besti[i], m, 32);
      bool take = (ov < bestv[i]) || (ov == bestv[i] && oi < besti[i]);
      bestv[i] = take ? ov : bestv[i];
      besti[i] = take ? oi : besti[i];
    }
  }

  // ---- Gather z_q, straight-through output, loss partial ------------------
  float lsum = 0.0f;
#pragma unroll
  for (int i = 0; i < 8; ++i) {
    const int grow = rowbase + h * 8 + i;
    const int idx  = besti[i];
    if (n == 0) out_idx[grow] = (float)idx;

    const float* erow = emb    + (size_t)idx  * VQ_D;
    const float* zr   = z      + (size_t)grow * VQ_D;
    float*       orow = out_zq + (size_t)grow * VQ_D;
#pragma unroll
    for (int t = 0; t < 4; ++t) {
      int d = n * 4 + t * 64;
      v4f e  = *(const v4f*)(erow + d);
      v4f zz = *(const v4f*)(zr + d);
      v4f df = e - zz;                    // z_q - z
      *(v4f*)(orow + d) = zz + df;        // z + (z_q - z), bit-faithful ST
      lsum += df[0] * df[0] + df[1] * df[1] + df[2] * df[2] + df[3] * df[3];
    }
  }
#pragma unroll
  for (int m = 16; m >= 1; m >>= 1) lsum += __shfl_xor(lsum, m, 32);
  if (lane == 0) {
    // loss = (1 + 0.25) * sum((z_q - z)^2) / (16*4096*256)
    unsafeAtomicAdd(loss_out, lsum * (1.25f / 16777216.0f));
  }
#undef VQ_ISSUE_CHUNK
}

// ---------------------------------------------------------------------------
extern "C" void kernel_launch(void* const* d_in, const int* in_sizes, int n_in,
                              void* d_out, int out_size, void* d_ws, size_t ws_size,
                              hipStream_t stream) {
  const float* z   = (const float*)d_in[0];   // [65536, 256] f32
  const float* emb = (const float*)d_in[1];   // [1024, 256] f32

  _Float16* emb16 = (_Float16*)d_ws;                                   // 512 KB
  float*    esq   = (float*)((char*)d_ws + (size_t)VQ_K * VQ_D * 2);   //   4 KB

  float* out_zq  = (float*)d_out;
  float* loss    = out_zq + (size_t)VQ_ROWS * VQ_D;
  float* out_idx = loss + 1;

  vq_prep_kernel<<<VQ_K / 8, 256, 0, stream>>>(emb, emb16, esq, loss);
  vq_main_kernel<<<VQ_ROWS / 128, 256, 0, stream>>>(z, emb, emb16, esq,
                                                    out_zq, loss, out_idx);
}
